// SpectralAttention_51170240365082
// MI455X (gfx1250) — compile-verified, matrix operand order checked by reading, not run on hardware
//
#include <hip/hip_runtime.h>
#include <hip/hip_bf16.h>

#define B_ 16
#define C_ 256
#define W_ 2048
#define ESTRIDE 2052   // 2048 + 4 floats pad (bank spread)
#define NBLK 8         // C_/32 K-dim blocks

typedef __attribute__((ext_vector_type(16))) __bf16 v16bf;
typedef __attribute__((ext_vector_type(8)))  float  v8f;

// CDNA5 16-bit operand K pattern (ISA 7.12.2):
// lanes 0-15: elements 0..7 -> K=0..7,  8..15 -> K=16..23
// lanes16-31: elements 0..7 -> K=8..15, 8..15 -> K=24..31
static __device__ __forceinline__ int kpat(int e, int half) {
  return (e & 7) + (half << 3) + ((e >> 3) << 4);
}
// inverse map: k-within-block kk -> (half, element)
static __device__ __forceinline__ int khalf(int kk) { return (kk >> 3) & 1; }
static __device__ __forceinline__ int kelem(int kk) {
  return (kk & 7) | (((kk >> 4) & 1) << 3);
}

static __device__ __forceinline__ v8f wmma_bf16(v16bf a, v16bf b, v8f c) {
  return __builtin_amdgcn_wmma_f32_16x16x32_bf16(false, a, false, b, (short)0, c,
                                                 false, false);
}

// ---------------- Kernel 0: one-shot weight conversion f32 -> bf16 ----------------
// Wq is pre-scaled by 1/sqrt(C) = 1/16 (exact power-of-two, no precision loss).
__global__ __launch_bounds__(256) void wconv_kernel(
    const float* __restrict__ Wq, const float* __restrict__ Wk,
    const float* __restrict__ Wv, __bf16* __restrict__ Wqb,
    __bf16* __restrict__ Wkb, __bf16* __restrict__ Wvb) {
  const int t = blockIdx.x * 256 + threadIdx.x;  // grid covers C_*C_ = 65536
  Wqb[t] = (__bf16)(Wq[t] * 0.0625f);
  Wkb[t] = (__bf16)Wk[t];
  Wvb[t] = (__bf16)Wv[t];
}

// ---------------- Kernel 1: K = Wk*x+bk (fragment layout), V = Wv*x+bv (natural) ----
// Kf layout: Kf[(((b*NBLK + blk)*2 + half)*W_ + j)*16 + e] holds
//            K[b][blk*32 + kpat(e,half)][j]  -> B-operand fragment = 32B/lane.
__global__ __launch_bounds__(256) void kv_gemm_kernel(
    const float* __restrict__ x, const __bf16* __restrict__ Wkb,
    const float* __restrict__ bk, const __bf16* __restrict__ Wvb,
    const float* __restrict__ bv, __bf16* __restrict__ Kf,
    __bf16* __restrict__ Vout) {
  __shared__ alignas(32) __bf16 xsw[NBLK * 32 * 16];   // x fragments [blk][lane][e]
  __shared__ alignas(32) __bf16 ksta[8][2][16][16];    // per-wave K D-tile staging
  const int b = blockIdx.y;
  const int n0 = blockIdx.x * 16;
  const int tid = threadIdx.x;
  const int w = tid >> 5, lane = tid & 31, lm = lane & 15, half = (lane >> 4) & 1;

  {  // stage x tile into fragment layout: x[c][n] -> xsw[blk][halfc*16+n][ec]
    const int blk = tid >> 5, hc = khalf(tid & 31), ec = kelem(tid & 31);
    const float* xp = x + ((size_t)b * C_ + tid) * W_ + n0;
#pragma unroll
    for (int n = 0; n < 16; ++n)
      xsw[((blk * 32 + hc * 16 + n) << 4) + ec] = (__bf16)xp[n];
  }
  __syncthreads();

  v16bf bxf[NBLK];  // shared B fragments, reused for both tiles and both K/V
#pragma unroll
  for (int blk = 0; blk < NBLK; ++blk)
    bxf[blk] = *(const v16bf*)(xsw + ((blk * 32 + lane) << 4));

#pragma unroll
  for (int t = 0; t < 2; ++t) {
    const int m0 = (w * 2 + t) * 16;  // output-channel tile (both t in c-block w)
    v8f accK = {}, accV = {};
#pragma unroll
    for (int blk = 0; blk < NBLK; ++blk) {
      v16bf aK, aV;
#pragma unroll
      for (int e = 0; e < 16; ++e) {
        const int k = blk * 32 + kpat(e, half);
        aK[e] = Wkb[(m0 + lm) * C_ + k];  // contiguous 16B runs -> b128 loads
        aV[e] = Wvb[(m0 + lm) * C_ + k];
      }
      accK = wmma_bf16(aK, bxf[blk], accK);
      accV = wmma_bf16(aV, bxf[blk], accV);
    }
#pragma unroll
    for (int r = 0; r < 8; ++r) {
      const int m = m0 + r + half * 8;
      Vout[((size_t)b * C_ + m) * W_ + n0 + lm] = (__bf16)(accV[r] + bv[m]);
      ksta[w][t][r + half * 8][lm] = (__bf16)(accK[r] + bk[m]);  // wave-local
    }
  }
  // merged K store: lane (half,j') emits one contiguous 32B fragment chunk
  {
    v16bf kfrag;
#pragma unroll
    for (int r = 0; r < 8; ++r) {
      kfrag[r]     = ksta[w][0][half * 8 + r][lm];  // ec = r   (kk<16)
      kfrag[8 + r] = ksta[w][1][half * 8 + r][lm];  // ec = r+8 (kk>=16)
    }
    const size_t kidx =
        ((((size_t)b * NBLK + w) * 2 + half) * W_ + n0 + lm) << 4;
    *(v16bf*)(Kf + kidx) = kfrag;  // lanes cover 512B contiguous
  }
}

// ---------------- Kernel 2: fused Q-proj + softmax(QK^T/sqrt(C)) + V*P^T ----------------
__global__ __launch_bounds__(256) void attn_kernel(
    const float* __restrict__ x, const __bf16* __restrict__ Wqb,
    const float* __restrict__ bq, const __bf16* __restrict__ Kf,
    const __bf16* __restrict__ Vb, const float* __restrict__ gamma,
    float* __restrict__ out) {
  extern __shared__ char smem[];
  float*  E    = (float*)smem;                    // [16][ESTRIDE] energy row-block
  __bf16* qTs  = (__bf16*)(E + 16 * ESTRIDE);     // qT fragments [blk][lane][e], 8KB
  __bf16* xsw  = qTs + NBLK * 32 * 16;            // x fragments  [blk][lane][e], 8KB
  float*  red  = (float*)(xsw + NBLK * 32 * 16);  // [16*16] reduction scratch
  float*  mrow = red + 256;                       // [16] row max
  float*  lrow = mrow + 16;                       // [16] row sum-exp

  const int b = blockIdx.y;
  const int i0 = blockIdx.x * 16;
  const int tid = threadIdx.x;
  const int w = tid >> 5, lane = tid & 31, lm = lane & 15, half = (lane >> 4) & 1;

  {  // stage x_i tile into fragment layout (coalesced 64B row reads)
    const int blk = tid >> 5, hc = khalf(tid & 31), ec = kelem(tid & 31);
    const float* xp = x + ((size_t)b * C_ + tid) * W_ + i0;
#pragma unroll
    for (int n = 0; n < 16; ++n)
      xsw[((blk * 32 + hc * 16 + n) << 4) + ec] = (__bf16)xp[n];
  }
  __syncthreads();

  // ---- Phase Q: qT = x_i^T * (Wq/16)^T + bq/16, written as A-fragments ----
  {
    v16bf xf[NBLK];
#pragma unroll
    for (int blk = 0; blk < NBLK; ++blk)
      xf[blk] = *(const v16bf*)(xsw + ((blk * 32 + lane) << 4));
#pragma unroll
    for (int t = 0; t < 2; ++t) {
      const int co0 = (w * 2 + t) * 16;  // c_out tile
      v8f acc = {};
#pragma unroll
      for (int blk = 0; blk < NBLK; ++blk) {
        v16bf bb;
#pragma unroll
        for (int e = 0; e < 16; ++e)
          bb[e] = Wqb[(co0 + lm) * C_ + blk * 32 + kpat(e, half)];
        acc = wmma_bf16(xf[blk], bb, acc);
      }
      const int c = co0 + lm;  // this lane's output channel (fixed)
      const int blkc = c >> 5, hc = khalf(c & 31), ec = kelem(c & 31);
      const float bias = bq[c] * 0.0625f;
#pragma unroll
      for (int r = 0; r < 8; ++r) {
        const int i = r + half * 8;
        qTs[((blkc * 32 + hc * 16 + i) << 4) + ec] = (__bf16)(acc[r] + bias);
      }
    }
  }
  __syncthreads();

  // ---- Phase E: E[i][j] = sum_c qT[i][c] * K[c][j]; qT fragments hoisted ----
  {
    v16bf afrag[NBLK];
#pragma unroll
    for (int blk = 0; blk < NBLK; ++blk)
      afrag[blk] = *(const v16bf*)(qTs + ((blk * 32 + lane) << 4));
    const __bf16* kbase = Kf + ((((size_t)b * NBLK) * 2 + half) * W_ + lm) * 16;
    for (int jb = 0; jb < 16; ++jb) {
      const int j0 = (jb * 8 + w) * 16;
      if (jb < 15) __builtin_prefetch(kbase + ((size_t)(j0 + 128) << 4), 0, 3);
      v8f acc = {};
#pragma unroll
      for (int blk = 0; blk < NBLK; ++blk) {
        const v16bf bb = *(const v16bf*)(kbase +
            (((size_t)blk * 2 * W_ + j0) << 4));  // 32B contiguous per lane
        acc = wmma_bf16(afrag[blk], bb, acc);
      }
#pragma unroll
      for (int r = 0; r < 8; ++r) E[(r + half * 8) * ESTRIDE + j0 + lm] = acc[r];
    }
  }
  __syncthreads();

  // ---- Phase S: exact softmax stats, in-place exp (16 threads per row) ----
  {
    const int row = tid >> 4, sub = tid & 15;
    float m = -3.4e38f;
    for (int j = sub; j < W_; j += 16) m = fmaxf(m, E[row * ESTRIDE + j]);
    red[row * 16 + sub] = m;
    __syncthreads();
    if (sub == 0) {
      float mm = red[row * 16];
      for (int s = 1; s < 16; ++s) mm = fmaxf(mm, red[row * 16 + s]);
      mrow[row] = mm;
    }
    __syncthreads();
    const float mm = mrow[row];
    float s = 0.f;
    for (int j = sub; j < W_; j += 16) {
      const float p = __expf(E[row * ESTRIDE + j] - mm);
      E[row * ESTRIDE + j] = p;
      s += p;
    }
    red[row * 16 + sub] = s;
    __syncthreads();
    if (sub == 0) {
      float ss = 0.f;
      for (int s2 = 0; s2 < 16; ++s2) ss += red[row * 16 + s2];
      lrow[row] = ss;
    }
  }
  __syncthreads();

  // ---- Phase O: out[c][i] = gamma * (sum_j V[c][j] P[i][j]) / l_i + x[c][i] ----
  const int cA = w * 16, cB = (w + 8) * 16;  // two c-tiles per wave
  const __bf16* vA = Vb + ((size_t)b * C_ + cA + lm) * W_;
  const __bf16* vB = Vb + ((size_t)b * C_ + cB + lm) * W_;
  v8f acc0 = {}, acc1 = {};
  for (int j0 = 0; j0 < W_; j0 += 32) {
    if (j0 + 32 < W_) {  // prefetch next V fragments
      __builtin_prefetch(vA + j0 + 32, 0, 3);
      __builtin_prefetch(vB + j0 + 32, 0, 3);
    }
    v16bf bb, a0, a1;
#pragma unroll
    for (int e = 0; e < 16; ++e) {
      const int k = j0 + kpat(e, half);
      bb[e] = (__bf16)E[lm * ESTRIDE + k];  // B: K=j, N=i (P^T); 32B f32 runs
      a0[e] = vA[k];                        // A: M=c, K=j; two 16B runs -> b128
      a1[e] = vB[k];
    }
    acc0 = wmma_bf16(a0, bb, acc0);
    acc1 = wmma_bf16(a1, bb, acc1);
  }
  const float g = gamma[0];
  const float linv = 1.0f / (lrow[lm] + 1e-10f);  // column i = lane&15
#pragma unroll
  for (int r = 0; r < 8; ++r) {
    const int off = r + half * 8;
    const size_t idx0 = ((size_t)b * C_ + cA + off) * W_ + i0 + lm;
    const size_t idx1 = ((size_t)b * C_ + cB + off) * W_ + i0 + lm;
    out[idx0] = g * acc0[r] * linv + x[idx0];
    out[idx1] = g * acc1[r] * linv + x[idx1];
  }
}

// ---------------- host launch ----------------
extern "C" void kernel_launch(void* const* d_in, const int* in_sizes, int n_in,
                              void* d_out, int out_size, void* d_ws, size_t ws_size,
                              hipStream_t stream) {
  const float* x     = (const float*)d_in[0];
  const float* Wq    = (const float*)d_in[1];
  const float* bq    = (const float*)d_in[2];
  const float* Wk    = (const float*)d_in[3];
  const float* bk    = (const float*)d_in[4];
  const float* Wv    = (const float*)d_in[5];
  const float* bv    = (const float*)d_in[6];
  const float* gamma = (const float*)d_in[7];
  float* out = (float*)d_out;

  __bf16* Kf  = (__bf16*)d_ws;                       // 16 MB bf16 (fragment layout)
  __bf16* Vb  = Kf + (size_t)B_ * C_ * W_;           // 16 MB bf16 (natural layout)
  __bf16* Wqb = Vb + (size_t)B_ * C_ * W_;           // 128 KB each
  __bf16* Wkb = Wqb + (size_t)C_ * C_;
  __bf16* Wvb = Wkb + (size_t)C_ * C_;

  wconv_kernel<<<(C_ * C_) / 256, 256, 0, stream>>>(Wq, Wk, Wv, Wqb, Wkb, Wvb);

  dim3 grid(W_ / 16, B_);
  kv_gemm_kernel<<<grid, 256, 0, stream>>>(x, Wkb, bk, Wvb, bv, Kf, Vb);

  const size_t lds_bytes = (size_t)16 * ESTRIDE * sizeof(float) +
                           (size_t)2 * NBLK * 32 * 16 * sizeof(__bf16) +
                           (256 + 16 + 16) * sizeof(float);  // ~145 KB (320 KB/WGP)
  attn_kernel<<<grid, 256, lds_bytes, stream>>>(x, Wqb, bq, Kf, Vb, gamma, out);
}
